// TemporalFusionTransformer_29222957482030
// MI455X (gfx1250) — matrix operand order repaired and checked by reading, not compiled, not run on hardware
//
#include <hip/hip_runtime.h>
#include <hip/hip_bf16.h>

typedef __attribute__((ext_vector_type(16))) _Float16 v16h;
typedef __attribute__((ext_vector_type(8)))  _Float16 v8h;
typedef __attribute__((ext_vector_type(4)))  _Float16 v4h;
typedef __attribute__((ext_vector_type(8)))  float    v8f;
typedef __attribute__((ext_vector_type(4)))  float    v4f;

#define DEV __device__ __forceinline__

DEV v8f vzero8() { v8f z; for (int i = 0; i < 8; ++i) z[i] = 0.f; return z; }
DEV float sigmoidf_(float x) { return 1.f / (1.f + __expf(-x)); }
DEV float eluf_(float x) { return x > 0.f ? x : (__expf(x) - 1.f); }

// A fragment for v_wmma_f32_16x16x32_f16 from a row-major f16 LDS tile.
// Caller passes k = kchunk + ((lane>>4)<<4); 16 contiguous halves.  Two 16B ds_load_b128 per lane.
DEV v16h frag_a_lds(const _Float16* base, int pitch, int row, int k) {
  const _Float16* p = base + row * pitch + k;
  v8h lo = *(const v8h*)(p);
  v8h hi = *(const v8h*)(p + 8);
  v16h r;
  for (int j = 0; j < 8; ++j) { r[j] = lo[j]; r[8 + j] = hi[j]; }
  return r;
}

// B fragment from an LDS tile stored transposed as BT[n][k] (pitch in halves).
// Caller passes k = kchunk + ((lane>>4)<<4); 16 contiguous halves.
DEV v16h frag_b_ldsT(const _Float16* base, int pitch, int col, int k) {
  const _Float16* p = base + col * pitch + k;
  v8h lo = *(const v8h*)(p);
  v8h hi = *(const v8h*)(p + 8);
  v16h r;
  for (int j = 0; j < 8; ++j) { r[j] = lo[j]; r[8 + j] = hi[j]; }
  return r;
}

DEV v8f wmma16(v16h a, v16h b, v8f c) {
  return __builtin_amdgcn_wmma_f32_16x16x32_f16(false, a, false, b, (short)0, c, false, false);
}

// ---------------------------------------------------------------------------
// Generic GEMM: C[M,N] = act(A16[M,K] @ B32[K,N] + bias)
// Block = 256 threads = 8 waves; block tile 32x128; BK = 64.
// Each wave owns two 16x16 output tiles -> 4 WMMAs per stage+barrier.
// ---------------------------------------------------------------------------
__global__ __launch_bounds__(256) void gemm_kernel(
    const _Float16* __restrict__ A, const float* __restrict__ B,
    const float* __restrict__ bias, float* __restrict__ C,
    _Float16* __restrict__ C16, int M, int N, int K, int act) {
  __shared__ _Float16 As[32 * 64];     // [m][k]   4 KB
  __shared__ _Float16 BTs[128 * 64];   // [n][k]  16 KB
  int tid = threadIdx.x, lane = tid & 31, wave = tid >> 5;
  int wm = wave >> 2, wn = wave & 3;
  int m0 = blockIdx.x * 32, n0 = blockIdx.y * 128;
  v8f acc0 = vzero8(), acc1 = vzero8();
  for (int k0 = 0; k0 < K; k0 += 64) {
    __syncthreads();
    {  // stage A 32x64 f16 (one 16B vector load/store per thread)
      int r = tid >> 3, c = (tid & 7) * 8;
      const v8h* s = (const v8h*)(A + (size_t)(m0 + r) * K + k0 + c);
      *(v8h*)&As[r * 64 + c] = *s;
      if (k0 + 64 < K) __builtin_prefetch((const void*)(s + 8), 0, 1);
    }
    {  // stage B 64x128 f32 -> f16 transposed
      int kr = tid >> 2, c0 = (tid & 3) * 32;
      const float* s = B + (size_t)(k0 + kr) * N + n0 + c0;
      if (k0 + 64 < K) __builtin_prefetch((const void*)(s + (size_t)64 * N), 0, 1);
      for (int i = 0; i < 32; ++i) BTs[(c0 + i) * 64 + kr] = (_Float16)s[i];
    }
    __syncthreads();
    for (int kc = 0; kc < 64; kc += 32) {
      v16h a = frag_a_lds(As, 64, wm * 16 + (lane & 15), kc + ((lane >> 4) << 4));
      acc0 = wmma16(a, frag_b_ldsT(BTs, 64, (wn * 2 + 0) * 16 + (lane & 15), kc + ((lane >> 4) << 4)), acc0);
      acc1 = wmma16(a, frag_b_ldsT(BTs, 64, (wn * 2 + 1) * 16 + (lane & 15), kc + ((lane >> 4) << 4)), acc1);
    }
  }
  int row = m0 + wm * 16 + ((lane >> 4) << 3);
  for (int t2 = 0; t2 < 2; ++t2) {
    int col = n0 + (wn * 2 + t2) * 16 + (lane & 15);
    float bv = bias ? bias[col] : 0.f;
    v8f acc = t2 ? acc1 : acc0;
    for (int r = 0; r < 8; ++r) {
      float v = acc[r] + bv;
      if (act == 1) v = eluf_(v);
      else if (act == 2) v = fmaxf(v, 0.f);
      size_t idx = (size_t)(row + r) * N + col;
      if (C) C[idx] = v;
      if (C16) C16[idx] = (_Float16)v;
    }
  }
}

// ---------------------------------------------------------------------------
// f32 -> f16 convert
// ---------------------------------------------------------------------------
__global__ void cvt_kernel(const float* __restrict__ s, _Float16* __restrict__ d, int n) {
  int i = blockIdx.x * blockDim.x + threadIdx.x;
  if (i < n) d[i] = (_Float16)s[i];
}

// ---------------------------------------------------------------------------
// Flat GRN tail: t = sigmoid(g)*o + skip ; LN(128) ; softmax(128)
// grid = 1024 rows, block = 128
// ---------------------------------------------------------------------------
__global__ __launch_bounds__(128) void grn_tail_kernel(
    const float* __restrict__ gb, const float* __restrict__ ob,
    const float* __restrict__ sk, const float* __restrict__ lng,
    const float* __restrict__ lnb, float* __restrict__ swb,
    float* __restrict__ out_sw) {
  __shared__ float sh[128];
  __shared__ float stat[2];
  int row = blockIdx.x, t = threadIdx.x;
  size_t idx = (size_t)row * 128 + t;
  float v = sigmoidf_(gb[idx]) * ob[idx] + sk[idx];
  sh[t] = v; __syncthreads();
  for (int st = 64; st > 0; st >>= 1) { if (t < st) sh[t] += sh[t + st]; __syncthreads(); }
  if (t == 0) stat[0] = sh[0] * (1.f / 128.f);
  __syncthreads();
  float d = v - stat[0];
  sh[t] = d * d; __syncthreads();
  for (int st = 64; st > 0; st >>= 1) { if (t < st) sh[t] += sh[t + st]; __syncthreads(); }
  if (t == 0) stat[1] = rsqrtf(sh[0] * (1.f / 128.f) + 1e-5f);
  __syncthreads();
  float ln = d * stat[1] * lng[t] + lnb[t];
  sh[t] = ln; __syncthreads();
  for (int st = 64; st > 0; st >>= 1) { if (t < st) sh[t] = fmaxf(sh[t], sh[t + st]); __syncthreads(); }
  if (t == 0) stat[0] = sh[0];
  __syncthreads();
  float e = __expf(ln - stat[0]);
  sh[t] = e; __syncthreads();
  for (int st = 64; st > 0; st >>= 1) { if (t < st) sh[t] += sh[t + st]; __syncthreads(); }
  if (t == 0) stat[1] = 1.f / sh[0];
  __syncthreads();
  float r = e * stat[1];
  swb[idx] = r;
  out_sw[idx] = r;
}

// ---------------------------------------------------------------------------
// Fused per-variable GRNs + variable selection (dominant compute, ~51 GFLOP).
// grid = 64 (16-row slabs of N=1024), block = 256 (8 waves cover all 256 cols).
// Loops f = 0..127; 3 WMMA GEMMs per feature with BK=64 staged weight chunks;
// per-variable LayerNorm + sw-weighted accumulation kept entirely in LDS.
// ---------------------------------------------------------------------------
__global__ __launch_bounds__(256) void sv_select_kernel(
    const float* __restrict__ x, const float* __restrict__ sw,
    const float* __restrict__ w1, const float* __restrict__ b1,
    const float* __restrict__ w2, const float* __restrict__ b2,
    const float* __restrict__ wo, const float* __restrict__ bo,
    const float* __restrict__ wg, const float* __restrict__ bg,
    const float* __restrict__ ssw, const float* __restrict__ ssb,
    const float* __restrict__ lng, const float* __restrict__ lnb,
    const float* __restrict__ pos, float* __restrict__ feats,
    _Float16* __restrict__ feats16) {
  __shared__ _Float16 Ah[16 * 256];    // h1 (f16)                     8 KB
  __shared__ _Float16 Hh[16 * 256];    // h2 (f16)                     8 KB
  __shared__ _Float16 BT[256 * 64];    // weight chunk [n][k]         32 KB
  __shared__ float Vb[16 * 256];       // pre-LN value                16 KB
  __shared__ float Sel[16 * 256];      // selection accumulator       16 KB
  __shared__ float redS[256], redQ[256];
  __shared__ float rowM[16], rowI[16];

  const int tid = threadIdx.x, lane = tid & 31, wave = tid >> 5;
  const int n0 = blockIdx.x * 16;
  const int rw = tid >> 4, cg = tid & 15;
  const int kr = tid >> 3, c0s = (tid & 7) * 32;  // weight staging coords

  for (int i = tid; i < 16 * 256; i += 256) Sel[i] = 0.f;

  for (int f = 0; f < 128; ++f) {
    __syncthreads();
    {  // h1 = elu(x[:,f] * w1[f] + b1[f])
      float xv = x[(size_t)(n0 + rw) * 128 + f];
      const float* w1f = w1 + f * 256;
      const float* b1f = b1 + f * 256;
      for (int i = 0; i < 16; ++i) {
        int c = cg * 16 + i;
        Ah[rw * 256 + c] = (_Float16)eluf_(xv * w1f[c] + b1f[c]);
      }
    }
    // GEMM1: h2 = elu(h1 @ w2[f] + b2[f])
    v8f a0 = vzero8(), a1 = vzero8();
    const float* W2 = w2 + (size_t)f * 65536;
    for (int k0 = 0; k0 < 256; k0 += 64) {
      __syncthreads();
      for (int rr = 0; rr < 64; rr += 32) {
        const float* s = W2 + (size_t)(k0 + kr + rr) * 256 + c0s;
        if (k0 + 64 < 256) __builtin_prefetch((const void*)(s + 64 * 256), 0, 1);
        for (int i = 0; i < 32; ++i) BT[(c0s + i) * 64 + kr + rr] = (_Float16)s[i];
      }
      __syncthreads();
      for (int kc = 0; kc < 64; kc += 32) {
        v16h a = frag_a_lds(Ah, 256, lane & 15, k0 + kc + ((lane >> 4) << 4));
        a0 = wmma16(a, frag_b_ldsT(BT, 64, (wave * 2 + 0) * 16 + (lane & 15), kc + ((lane >> 4) << 4)), a0);
        a1 = wmma16(a, frag_b_ldsT(BT, 64, (wave * 2 + 1) * 16 + (lane & 15), kc + ((lane >> 4) << 4)), a1);
      }
    }
    __syncthreads();
    { int mr = (lane >> 4) * 8;
      const float* b2f = b2 + f * 256;
      int c0 = (wave * 2 + 0) * 16 + (lane & 15);
      int c1 = (wave * 2 + 1) * 16 + (lane & 15);
      for (int r = 0; r < 8; ++r) {
        Hh[(mr + r) * 256 + c0] = (_Float16)eluf_(a0[r] + b2f[c0]);
        Hh[(mr + r) * 256 + c1] = (_Float16)eluf_(a1[r] + b2f[c1]);
      } }
    __syncthreads();
    // GEMM2: vo = h2 @ wo[f]
    v8f vo0 = vzero8(), vo1 = vzero8();
    const float* WO = wo + (size_t)f * 65536;
    for (int k0 = 0; k0 < 256; k0 += 64) {
      __syncthreads();
      for (int rr = 0; rr < 64; rr += 32) {
        const float* s = WO + (size_t)(k0 + kr + rr) * 256 + c0s;
        if (k0 + 64 < 256) __builtin_prefetch((const void*)(s + 64 * 256), 0, 1);
        for (int i = 0; i < 32; ++i) BT[(c0s + i) * 64 + kr + rr] = (_Float16)s[i];
      }
      __syncthreads();
      for (int kc = 0; kc < 64; kc += 32) {
        v16h a = frag_a_lds(Hh, 256, lane & 15, k0 + kc + ((lane >> 4) << 4));
        vo0 = wmma16(a, frag_b_ldsT(BT, 64, (wave * 2 + 0) * 16 + (lane & 15), kc + ((lane >> 4) << 4)), vo0);
        vo1 = wmma16(a, frag_b_ldsT(BT, 64, (wave * 2 + 1) * 16 + (lane & 15), kc + ((lane >> 4) << 4)), vo1);
      }
    }
    // GEMM3: vg = h2 @ wg[f]
    v8f vg0 = vzero8(), vg1 = vzero8();
    const float* WG = wg + (size_t)f * 65536;
    for (int k0 = 0; k0 < 256; k0 += 64) {
      __syncthreads();
      for (int rr = 0; rr < 64; rr += 32) {
        const float* s = WG + (size_t)(k0 + kr + rr) * 256 + c0s;
        if (k0 + 64 < 256) __builtin_prefetch((const void*)(s + 64 * 256), 0, 1);
        for (int i = 0; i < 32; ++i) BT[(c0s + i) * 64 + kr + rr] = (_Float16)s[i];
      }
      __syncthreads();
      for (int kc = 0; kc < 64; kc += 32) {
        v16h a = frag_a_lds(Hh, 256, lane & 15, k0 + kc + ((lane >> 4) << 4));
        vg0 = wmma16(a, frag_b_ldsT(BT, 64, (wave * 2 + 0) * 16 + (lane & 15), kc + ((lane >> 4) << 4)), vg0);
        vg1 = wmma16(a, frag_b_ldsT(BT, 64, (wave * 2 + 1) * 16 + (lane & 15), kc + ((lane >> 4) << 4)), vg1);
      }
    }
    __syncthreads();
    {  // v = sigmoid(vg+bg)*(vo+bo) + (x[:,f]*sv_sw[f] + sv_sb[f])
      int mr = (lane >> 4) * 8;
      const float* bof = bo + f * 256;
      const float* bgf = bg + f * 256;
      const float* swf = ssw + f * 256;
      const float* sbf = ssb + f * 256;
      for (int t2 = 0; t2 < 2; ++t2) {
        int cc = (wave * 2 + t2) * 16 + (lane & 15);
        for (int r = 0; r < 8; ++r) {
          int m = mr + r;
          float xv = x[(size_t)(n0 + m) * 128 + f];
          float oo = (t2 ? vo1[r] : vo0[r]) + bof[cc];
          float gg = sigmoidf_((t2 ? vg1[r] : vg0[r]) + bgf[cc]);
          Vb[m * 256 + cc] = gg * oo + xv * swf[cc] + sbf[cc];
        }
      }
    }
    __syncthreads();
    {  // LN stats per row (H=256)
      float s = 0.f, q = 0.f;
      for (int i = 0; i < 16; ++i) { float v = Vb[rw * 256 + cg * 16 + i]; s += v; q += v * v; }
      redS[tid] = s; redQ[tid] = q;
    }
    __syncthreads();
    if (cg == 0) {
      float S = 0.f, Q = 0.f;
      for (int i = 0; i < 16; ++i) { S += redS[rw * 16 + i]; Q += redQ[rw * 16 + i]; }
      float mean = S * (1.f / 256.f);
      float var = Q * (1.f / 256.f) - mean * mean;
      rowM[rw] = mean; rowI[rw] = rsqrtf(var + 1e-5f);
    }
    __syncthreads();
    {  // Sel += sw[:,f] * LN(v)
      float swv = sw[(size_t)(n0 + rw) * 128 + f];
      float mean = rowM[rw], inv = rowI[rw];
      const float* lg = lng + f * 256;
      const float* lb = lnb + f * 256;
      for (int i = 0; i < 16; ++i) {
        int c = cg * 16 + i;
        float v = (Vb[rw * 256 + c] - mean) * inv * lg[c] + lb[c];
        Sel[rw * 256 + c] += swv * v;
      }
    }
  }
  __syncthreads();
  for (int i = tid; i < 16 * 256; i += 256) {
    int m = i >> 8, c = i & 255;
    int n = n0 + m, s = n & 127;
    float v = Sel[i] + pos[s * 256 + c];
    feats[(size_t)n * 256 + c] = v;
    feats16[(size_t)n * 256 + c] = (_Float16)v;
  }
}

// ---------------------------------------------------------------------------
// Persistent single-WG WMMA LSTM: gates[16,1024] = [x|h][16,512] @ [wih;whh]^T
// h/c state and gate scratch in LDS; 128 sequential steps.
// ---------------------------------------------------------------------------
__global__ __launch_bounds__(256) void lstm_kernel(
    const float* __restrict__ xin, const float* __restrict__ wih,
    const float* __restrict__ whh, const float* __restrict__ bih,
    const float* __restrict__ bhh, float* __restrict__ out) {
  __shared__ _Float16 Ah[16 * 512];  // rows 0-7 = batch, cols 0-255 x_t, 256-511 h
  __shared__ float Gb[8 * 1024];
  __shared__ float Cst[8 * 256];
  int tid = threadIdx.x, lane = tid & 31, wave = tid >> 5;
  for (int i = tid; i < 16 * 512; i += 256) Ah[i] = (_Float16)0.f;
  for (int i = tid; i < 8 * 256; i += 256) Cst[i] = 0.f;
  __syncthreads();
  for (int s = 0; s < 128; ++s) {
    { int b = tid >> 5, c0 = (tid & 31) * 8;
      const float* src = xin + (size_t)(b * 128 + s) * 256 + c0;
      for (int i = 0; i < 8; ++i) Ah[b * 512 + c0 + i] = (_Float16)src[i]; }
    __syncthreads();
    v8f acc[8];
    for (int t = 0; t < 8; ++t) acc[t] = vzero8();
    for (int k0 = 0; k0 < 512; k0 += 32) {
      v16h a = frag_a_lds(Ah, 512, lane & 15, k0 + ((lane >> 4) << 4));
      int ks = k0 + ((lane >> 4) << 4);
      for (int t = 0; t < 8; ++t) {
        int n = (wave * 8 + t) * 16 + (lane & 15);
        const float* src = (ks < 256) ? (wih + (size_t)n * 256 + ks)
                                      : (whh + (size_t)n * 256 + (ks - 256));
        const v4f* s4 = (const v4f*)src;
        v4f q0 = s4[0], q1 = s4[1], q2 = s4[2], q3 = s4[3];
        v16h bf;
        for (int j = 0; j < 4; ++j) {
          bf[j] = (_Float16)q0[j]; bf[4 + j] = (_Float16)q1[j];
          bf[8 + j] = (_Float16)q2[j]; bf[12 + j] = (_Float16)q3[j];
        }
        acc[t] = wmma16(a, bf, acc[t]);
      }
    }
    { int mr = (lane >> 4) * 8;
      for (int t = 0; t < 8; ++t) {
        int cc = (wave * 8 + t) * 16 + (lane & 15);
        float bv = bih[cc] + bhh[cc];
        for (int r = 0; r < 8; ++r) { int m = mr + r; if (m < 8) Gb[m * 1024 + cc] = acc[t][r] + bv; }
      } }
    __syncthreads();
    { int u = tid;  // hidden unit
      for (int b = 0; b < 8; ++b) {
        float ig = Gb[b * 1024 + u], fg = Gb[b * 1024 + 256 + u];
        float gg = Gb[b * 1024 + 512 + u], og = Gb[b * 1024 + 768 + u];
        float c = sigmoidf_(fg) * Cst[b * 256 + u] + sigmoidf_(ig) * tanhf(gg);
        float h = sigmoidf_(og) * tanhf(c);
        Cst[b * 256 + u] = c;
        Ah[b * 512 + 256 + u] = (_Float16)h;
        out[(size_t)(b * 128 + s) * 256 + u] = h;
      } }
    __syncthreads();
  }
}

// ---------------------------------------------------------------------------
// Attention core per (batch, head): scores = Q K^T * scale, softmax, ctx = P V
// grid = 64 = B*NH, block = 256.
// ---------------------------------------------------------------------------
__global__ __launch_bounds__(256) void attn_kernel(
    const float* __restrict__ Q, const float* __restrict__ K,
    const float* __restrict__ V, _Float16* __restrict__ ctx16) {
  __shared__ _Float16 Qs[128 * 32];
  __shared__ _Float16 Ks[128 * 32];   // natural [s][d] == B-transposed layout
  __shared__ _Float16 VT[32 * 128];   // [d][s]
  __shared__ float Sc[128 * 128];
  __shared__ _Float16 Ps[128 * 128];
  int tid = threadIdx.x, lane = tid & 31, wave = tid >> 5;
  int b = blockIdx.x >> 3, h = blockIdx.x & 7, ho = h * 32;
  { int r = tid >> 1, c0 = (tid & 1) * 16;
    const float* q = Q + (size_t)(b * 128 + r) * 256 + ho + c0;
    const float* k = K + (size_t)(b * 128 + r) * 256 + ho + c0;
    const float* v = V + (size_t)(b * 128 + r) * 256 + ho + c0;
    for (int i = 0; i < 16; ++i) {
      Qs[r * 32 + c0 + i] = (_Float16)q[i];
      Ks[r * 32 + c0 + i] = (_Float16)k[i];
      VT[(c0 + i) * 128 + r] = (_Float16)v[i];
    } }
  __syncthreads();
  const float scale = 0.17677669529663687f;  // 1/sqrt(32)
  { v16h a = frag_a_lds(Qs, 32, wave * 16 + (lane & 15), (lane >> 4) << 4);
    for (int kt = 0; kt < 8; ++kt) {
      v16h bf = frag_b_ldsT(Ks, 32, kt * 16 + (lane & 15), (lane >> 4) << 4);
      v8f acc = vzero8();
      acc = wmma16(a, bf, acc);
      int m = wave * 16 + ((lane >> 4) << 3), n = kt * 16 + (lane & 15);
      for (int r = 0; r < 8; ++r) Sc[(m + r) * 128 + n] = acc[r] * scale;
    } }
  __syncthreads();
  if (tid < 128) {
    float mx = -1e30f;
    for (int c = 0; c < 128; ++c) mx = fmaxf(mx, Sc[tid * 128 + c]);
    float sum = 0.f;
    for (int c = 0; c < 128; ++c) { float e = __expf(Sc[tid * 128 + c] - mx); sum += e; Sc[tid * 128 + c] = e; }
    float inv = 1.f / sum;
    for (int c = 0; c < 128; ++c) Ps[tid * 128 + c] = (_Float16)(Sc[tid * 128 + c] * inv);
  }
  __syncthreads();
  for (int t = 0; t < 2; ++t) {
    int id = wave * 2 + t, mt = id >> 1, nt = id & 1;
    v8f acc = vzero8();
    for (int k0 = 0; k0 < 128; k0 += 32) {
      v16h a = frag_a_lds(Ps, 128, mt * 16 + (lane & 15), k0 + ((lane >> 4) << 4));
      v16h bf = frag_b_ldsT(VT, 128, nt * 16 + (lane & 15), k0 + ((lane >> 4) << 4));
      acc = wmma16(a, bf, acc);
    }
    int m = mt * 16 + ((lane >> 4) << 3), n = nt * 16 + (lane & 15);
    for (int r = 0; r < 8; ++r)
      ctx16[(size_t)(b * 128 + m + r) * 256 + ho + n] = (_Float16)acc[r];
  }
}

// ---------------------------------------------------------------------------
// a = LN(proj + resid); writes f32 (in place over resid buffer) and f16 copy.
// ---------------------------------------------------------------------------
__global__ __launch_bounds__(256) void resid_ln_kernel(
    const float* __restrict__ proj, const float* __restrict__ resid,
    const float* __restrict__ g, const float* __restrict__ bta,
    float* __restrict__ outf, _Float16* __restrict__ out16) {
  __shared__ float sh[256];
  __shared__ float stat[2];
  int row = blockIdx.x, t = threadIdx.x;
  size_t idx = (size_t)row * 256 + t;
  float v = proj[idx] + resid[idx];
  sh[t] = v; __syncthreads();
  for (int st = 128; st > 0; st >>= 1) { if (t < st) sh[t] += sh[t + st]; __syncthreads(); }
  if (t == 0) stat[0] = sh[0] * (1.f / 256.f);
  __syncthreads();
  float d = v - stat[0];
  sh[t] = d * d; __syncthreads();
  for (int st = 128; st > 0; st >>= 1) { if (t < st) sh[t] += sh[t + st]; __syncthreads(); }
  if (t == 0) stat[1] = rsqrtf(sh[0] * (1.f / 256.f) + 1e-5f);
  __syncthreads();
  float o = d * stat[1] * g[t] + bta[t];
  outf[idx] = o;
  out16[idx] = (_Float16)o;
}

// ---------------------------------------------------------------------------
// Heads: gctx = mean_s(a); pf = relu(gctx W1 + b1) W2 + b2 ; q10/50/90.
// grid = 8 (batch), block = 256.
// ---------------------------------------------------------------------------
__global__ __launch_bounds__(256) void head_kernel(
    const float* __restrict__ a, const float* __restrict__ w1,
    const float* __restrict__ b1, const float* __restrict__ w2,
    const float* __restrict__ b2, const float* __restrict__ qw,
    const float* __restrict__ qb, float* __restrict__ out) {
  __shared__ float g[256];
  __shared__ float hb[128];
  int b = blockIdx.x, t = threadIdx.x;
  float s = 0.f;
  for (int r = 0; r < 128; ++r) s += a[(size_t)(b * 128 + r) * 256 + t];
  g[t] = s * (1.f / 128.f);
  __syncthreads();
  if (t < 128) {
    float acc = b1[t];
    for (int k = 0; k < 256; ++k) acc += g[k] * w1[k * 128 + t];
    hb[t] = fmaxf(acc, 0.f);
  }
  __syncthreads();
  if (t < 24) {
    float acc = b2[t];
    for (int k = 0; k < 128; ++k) acc += hb[k] * w2[k * 24 + t];
    out[b * 24 + t] = acc;
  }
  if (t >= 32 && t < 56) {
    int j = t - 32; float acc = qb[j];
    for (int k = 0; k < 256; ++k) acc += g[k] * qw[k * 24 + j];
    out[192 + b * 24 + j] = acc;
  }
  if (t >= 64 && t < 88) {
    int j = t - 64; float acc = qb[24 + j];
    for (int k = 0; k < 256; ++k) acc += g[k] * qw[6144 + k * 24 + j];
    out[384 + b * 24 + j] = acc;
  }
  if (t >= 96 && t < 120) {
    int j = t - 96; float acc = qb[48 + j];
    for (int k = 0; k < 256; ++k) acc += g[k] * qw[12288 + k * 24 + j];
    out[576 + b * 24 + j] = acc;
  }
}

// ---------------------------------------------------------------------------
extern "C" void kernel_launch(void* const* d_in, const int* in_sizes, int n_in,
                              void* d_out, int out_size, void* d_ws, size_t ws_size,
                              hipStream_t stream) {
  (void)in_sizes; (void)n_in; (void)out_size; (void)ws_size;
  const float* x      = (const float*)d_in[0];
  const float* fg_w1  = (const float*)d_in[1];
  const float* fg_b1  = (const float*)d_in[2];
  const float* fg_w2  = (const float*)d_in[3];
  const float* fg_b2  = (const float*)d_in[4];
  const float* fg_wo  = (const float*)d_in[5];
  const float* fg_bo  = (const float*)d_in[6];
  const float* fg_wg  = (const float*)d_in[7];
  const float* fg_bg  = (const float*)d_in[8];
  const float* fg_sw  = (const float*)d_in[9];
  const float* fg_sb  = (const float*)d_in[10];
  const float* fg_lng = (const float*)d_in[11];
  const float* fg_lnb = (const float*)d_in[12];
  const float* sv_w1  = (const float*)d_in[13];
  const float* sv_b1  = (const float*)d_in[14];
  const float* sv_w2  = (const float*)d_in[15];
  const float* sv_b2  = (const float*)d_in[16];
  const float* sv_wo  = (const float*)d_in[17];
  const float* sv_bo  = (const float*)d_in[18];
  const float* sv_wg  = (const float*)d_in[19];
  const float* sv_bg  = (const float*)d_in[20];
  const float* sv_sw  = (const float*)d_in[21];
  const float* sv_sb  = (const float*)d_in[22];
  const float* sv_lng = (const float*)d_in[23];
  const float* sv_lnb = (const float*)d_in[24];
  const float* pos    = (const float*)d_in[25];
  const float* l0_wih = (const float*)d_in[26];
  const float* l0_whh = (const float*)d_in[27];
  const float* l0_bih = (const float*)d_in[28];
  const float* l0_bhh = (const float*)d_in[29];
  const float* l1_wih = (const float*)d_in[30];
  const float* l1_whh = (const float*)d_in[31];
  const float* l1_bih = (const float*)d_in[32];
  const float* l1_bhh = (const float*)d_in[33];
  const float* att_wq = (const float*)d_in[34];
  const float* att_wk = (const float*)d_in[35];
  const float* att_wv = (const float*)d_in[36];
  const float* att_wo = (const float*)d_in[37];
  const float* att_bo = (const float*)d_in[38];
  const float* att_lng= (const float*)d_in[39];
  const float* att_lnb= (const float*)d_in[40];
  const float* out_w1 = (const float*)d_in[41];
  const float* out_b1 = (const float*)d_in[42];
  const float* out_w2 = (const float*)d_in[43];
  const float* out_b2 = (const float*)d_in[44];
  const float* qw     = (const float*)d_in[45];
  const float* qb     = (const float*)d_in[46];
  float* out = (float*)d_out;

  char* base = (char*)d_ws;
  size_t off = 0;
  auto alloc = [&](size_t bytes) -> void* {
    off = (off + 255) & ~(size_t)255;
    void* r = base + off;
    off += bytes;
    return r;
  };
  _Float16* xf16    = (_Float16*)alloc(1024 * 128 * 2);
  _Float16* t0h     = (_Float16*)alloc(1024 * 256 * 2);
  _Float16* t1h     = (_Float16*)alloc(1024 * 256 * 2);
  float*    o_buf   = (float*)alloc(1024 * 128 * 4);
  float*    g_buf   = (float*)alloc(1024 * 128 * 4);
  float*    s_buf   = (float*)alloc(1024 * 128 * 4);
  float*    swb     = (float*)alloc(1024 * 128 * 4);
  float*    feats   = (float*)alloc(1024 * 256 * 4);
  _Float16* feats16 = (_Float16*)alloc(1024 * 256 * 2);
  float*    a0      = (float*)alloc(1024 * 256 * 4);
  float*    a1      = (float*)alloc(1024 * 256 * 4);
  _Float16* a16     = (_Float16*)alloc(1024 * 256 * 2);
  float*    qbf     = (float*)alloc(1024 * 256 * 4);
  float*    kbf     = (float*)alloc(1024 * 256 * 4);
  float*    vbf     = (float*)alloc(1024 * 256 * 4);
  _Float16* ctx16   = (_Float16*)alloc(1024 * 256 * 2);
  float*    proj    = (float*)alloc(1024 * 256 * 4);

  // ---- flattened GRN -> variable selection weights ----
  cvt_kernel<<<512, 256, 0, stream>>>(x, xf16, 1024 * 128);
  gemm_kernel<<<dim3(32, 2), 256, 0, stream>>>(xf16, fg_w1, fg_b1, nullptr, t0h, 1024, 256, 128, 1);
  gemm_kernel<<<dim3(32, 2), 256, 0, stream>>>(t0h, fg_w2, fg_b2, nullptr, t1h, 1024, 256, 256, 1);
  gemm_kernel<<<dim3(32, 1), 256, 0, stream>>>(t1h, fg_wo, fg_bo, o_buf, nullptr, 1024, 128, 256, 0);
  gemm_kernel<<<dim3(32, 1), 256, 0, stream>>>(t1h, fg_wg, fg_bg, g_buf, nullptr, 1024, 128, 256, 0);
  gemm_kernel<<<dim3(32, 1), 256, 0, stream>>>(xf16, fg_sw, fg_sb, s_buf, nullptr, 1024, 128, 128, 0);
  grn_tail_kernel<<<1024, 128, 0, stream>>>(g_buf, o_buf, s_buf, fg_lng, fg_lnb, swb, out + 768);

  // ---- fused per-variable GRNs + selection (dominant compute) ----
  sv_select_kernel<<<64, 256, 0, stream>>>(x, swb, sv_w1, sv_b1, sv_w2, sv_b2,
                                           sv_wo, sv_bo, sv_wg, sv_bg, sv_sw, sv_sb,
                                           sv_lng, sv_lnb, pos, feats, feats16);

  // ---- 2-layer LSTM (sequential, persistent WG) ----
  lstm_kernel<<<1, 256, 0, stream>>>(feats, l0_wih, l0_whh, l0_bih, l0_bhh, a0);
  lstm_kernel<<<1, 256, 0, stream>>>(a0, l1_wih, l1_whh, l1_bih, l1_bhh, a1);
  cvt_kernel<<<1024, 256, 0, stream>>>(a1, a16, 1024 * 256);

  // ---- NL attention layers ----
  for (int l = 0; l < 4; ++l) {
    const float* wq = att_wq + (size_t)l * 65536;
    const float* wk = att_wk + (size_t)l * 65536;
    const float* wv = att_wv + (size_t)l * 65536;
    const float* wo = att_wo + (size_t)l * 65536;
    const float* bo = att_bo + (size_t)l * 256;
    const float* lg = att_lng + (size_t)l * 256;
    const float* lb = att_lnb + (size_t)l * 256;
    gemm_kernel<<<dim3(32, 2), 256, 0, stream>>>(a16, wq, nullptr, qbf, nullptr, 1024, 256, 256, 0);
    gemm_kernel<<<dim3(32, 2), 256, 0, stream>>>(a16, wk, nullptr, kbf, nullptr, 1024, 256, 256, 0);
    gemm_kernel<<<dim3(32, 2), 256, 0, stream>>>(a16, wv, nullptr, vbf, nullptr, 1024, 256, 256, 0);
    attn_kernel<<<64, 256, 0, stream>>>(qbf, kbf, vbf, ctx16);
    gemm_kernel<<<dim3(32, 2), 256, 0, stream>>>(ctx16, wo, bo, proj, nullptr, 1024, 256, 256, 0);
    resid_ln_kernel<<<1024, 256, 0, stream>>>(proj, a1, lg, lb, a1, a16);
  }

  // ---- outputs: a, heads ----
  hipMemcpyAsync(out + 131840, a1, 1024 * 256 * sizeof(float),
                 hipMemcpyDeviceToDevice, stream);
  head_kernel<<<8, 256, 0, stream>>>(a1, out_w1, out_b1, out_w2, out_b2, qw, qb, out);
}
